// KDimSelfAttention_5497558138987
// MI455X (gfx1250) — matrix operand order repaired and checked by reading, not compiled, not run on hardware
//
#include <hip/hip_runtime.h>

typedef __attribute__((ext_vector_type(16))) _Float16 v16h;
typedef __attribute__((ext_vector_type(8)))  float    v8f;

#define T_DIM 2048
#define K_DIM 64

// ---------------------------------------------------------------------------
// Kernel A: column-sums of Wq/Wk and bias sums.
// misc layout (floats): [0..63]=wq_colsum, [64..127]=wk_colsum,
//                       [128]=sum(bq), [129]=sum(bk),
//                       [160..191]=ks_max per bc, [192..223]=ks_min per bc
// ---------------------------------------------------------------------------
__global__ void prep_weights(const float* __restrict__ Wq, const float* __restrict__ bq,
                             const float* __restrict__ Wk, const float* __restrict__ bk,
                             float* __restrict__ misc) {
  int k = threadIdx.x;
  if (k < K_DIM) {
    float sq = 0.f, sk = 0.f;
    for (int d = 0; d < K_DIM; ++d) { sq += Wq[d * K_DIM + k]; sk += Wk[d * K_DIM + k]; }
    misc[k] = sq;
    misc[K_DIM + k] = sk;
  }
  if (k == 0) {
    float s0 = 0.f, s1 = 0.f;
    for (int d = 0; d < K_DIM; ++d) { s0 += bq[d]; s1 += bk[d]; }
    misc[128] = s0;
    misc[129] = s1;
  }
}

// ---------------------------------------------------------------------------
// Kernel B: per row t compute a_t = scale*qs[t], ks[t], and V row -> Vt (f16,
// transposed to [bc][d][t] so WMMA B fragments are K-contiguous loads).
// ---------------------------------------------------------------------------
__global__ void proj_kernel(const float* __restrict__ x, const float* __restrict__ Wv,
                            const float* __restrict__ bv, const float* __restrict__ misc,
                            float* __restrict__ a_arr, float* __restrict__ ks_arr,
                            _Float16* __restrict__ Vt) {
  int gt = blockIdx.x * 64 + threadIdx.x;        // 0 .. B*C*T-1
  int bc = gt >> 11;                             // T = 2048
  int t  = gt & (T_DIM - 1);

  const float4* xp = (const float4*)(x + (size_t)gt * K_DIM);
  float4 xr[16];
#pragma unroll
  for (int j = 0; j < 16; ++j) xr[j] = xp[j];

  const float4* wq4 = (const float4*)misc;
  const float4* wk4 = (const float4*)(misc + K_DIM);
  float aq = misc[128], ak = misc[129];
#pragma unroll
  for (int j = 0; j < 16; ++j) {
    float4 q = wq4[j], kk = wk4[j], xv = xr[j];
    aq += xv.x * q.x + xv.y * q.y + xv.z * q.z + xv.w * q.w;
    ak += xv.x * kk.x + xv.y * kk.y + xv.z * kk.z + xv.w * kk.w;
  }
  a_arr[gt]  = aq * 0.125f;                      // 1/sqrt(64) folded in
  ks_arr[gt] = ak;

  for (int d = 0; d < K_DIM; ++d) {
    const float4* wv4 = (const float4*)(Wv + d * K_DIM);
    float acc = bv[d];
#pragma unroll
    for (int j = 0; j < 16; ++j) {
      float4 w = wv4[j], xv = xr[j];
      acc += xv.x * w.x + xv.y * w.y + xv.z * w.z + xv.w * w.w;
    }
    Vt[((size_t)bc * K_DIM + d) * T_DIM + t] = (_Float16)acc;
  }
}

// ---------------------------------------------------------------------------
// Kernel C: per-(b,c) max/min of ks (row-max of the rank-1 score matrix is
// a_t * ks_max or a_t * ks_min depending on sign(a_t)).
// ---------------------------------------------------------------------------
__global__ void minmax_kernel(const float* __restrict__ ks_arr, float* __restrict__ misc) {
  __shared__ float smax[256], smin[256];
  int bc = blockIdx.x, tid = threadIdx.x;
  const float* p = ks_arr + (size_t)bc * T_DIM;
  float mx = -1e30f, mn = 1e30f;
  for (int i = tid; i < T_DIM; i += 256) {
    float v = p[i];
    mx = fmaxf(mx, v);
    mn = fminf(mn, v);
  }
  smax[tid] = mx; smin[tid] = mn;
  __syncthreads();
  for (int s = 128; s > 0; s >>= 1) {
    if (tid < s) {
      smax[tid] = fmaxf(smax[tid], smax[tid + s]);
      smin[tid] = fminf(smin[tid], smin[tid + s]);
    }
    __syncthreads();
  }
  if (tid == 0) { misc[160 + bc] = smax[0]; misc[192 + bc] = smin[0]; }
}

// ---------------------------------------------------------------------------
// Kernel D: fused softmax(rank-1 scores) @ V via v_wmma_f32_16x16x32_f16.
// One wave32 per 16-row output tile; 64 s-chunks of 32; 4 N-tiles of 16.
// A-fragment (16x32 f16): lane L<16 owns row M=L, K offsets 0..15;
// lanes 16..31 own K offsets 16..31 (contiguous, same order as B).
// B-fragment (32x16 f16): lane n<16 = column n with K=0..15 contiguous,
// lanes 16..31 hold K=16..31 -> one 32B load from Vt per fragment.
// Denominator: 4 independent partial accumulators so the fp32 adds pipeline
// under the WMMAs instead of forming a 16-deep VALU_DEP_1 chain.
// ---------------------------------------------------------------------------
__global__ void attn_kernel(const float* __restrict__ a_arr, const float* __restrict__ ks_arr,
                            const float* __restrict__ misc, const _Float16* __restrict__ Vt,
                            float* __restrict__ out) {
  __shared__ float denom_lds[16];
  int tile = blockIdx.x;               // 0..127  (T/16)
  int bc   = blockIdx.y;               // 0..31   (B*C)
  int lane = threadIdx.x;              // 0..31
  int tbase = tile * 16;
  int lrow = lane & 15;
  int hi   = (lane >> 4) & 1;          // 0: lanes 0-15, 1: lanes 16-31

  float at   = a_arr[(size_t)bc * T_DIM + tbase + lrow];
  float kmax = misc[160 + bc], kmin = misc[192 + bc];
  float m    = (at >= 0.f) ? at * kmax : at * kmin;   // exact row max

  const float4*    ks4   = (const float4*)(ks_arr + (size_t)bc * T_DIM);
  const _Float16*  vbase = Vt + (size_t)bc * K_DIM * T_DIM;

  v8f c0 = {}, c1 = {}, c2 = {}, c3 = {};
  float ds0 = 0.f, ds1 = 0.f, ds2 = 0.f, ds3 = 0.f;

  for (int ch = 0; ch < 64; ++ch) {
    int b4 = ch * 8 + hi * 4;
    float4 q0 = ks4[b4], q1 = ks4[b4 + 1], q2 = ks4[b4 + 2], q3 = ks4[b4 + 3];
    float kv[16] = { q0.x, q0.y, q0.z, q0.w, q1.x, q1.y, q1.z, q1.w,
                     q2.x, q2.y, q2.z, q2.w, q3.x, q3.y, q3.z, q3.w };
    float e[16];
#pragma unroll
    for (int h = 0; h < 16; ++h) e[h] = __expf(at * kv[h] - m);

    v16h af;
#pragma unroll
    for (int h = 0; h < 16; ++h) af[h] = (_Float16)e[h];

    // 4 independent accumulation chains (explicit: fp adds don't reassociate)
#pragma unroll
    for (int h = 0; h < 16; h += 4) {
      ds0 += e[h + 0];
      ds1 += e[h + 1];
      ds2 += e[h + 2];
      ds3 += e[h + 3];
    }

    int soff = ch * 32 + hi * 16;
    v16h b0 = *(const v16h*)(vbase + (size_t)( 0 + lrow) * T_DIM + soff);
    v16h b1 = *(const v16h*)(vbase + (size_t)(16 + lrow) * T_DIM + soff);
    v16h b2 = *(const v16h*)(vbase + (size_t)(32 + lrow) * T_DIM + soff);
    v16h b3 = *(const v16h*)(vbase + (size_t)(48 + lrow) * T_DIM + soff);

    c0 = __builtin_amdgcn_wmma_f32_16x16x32_f16(false, af, false, b0, (short)0, c0, false, false);
    c1 = __builtin_amdgcn_wmma_f32_16x16x32_f16(false, af, false, b1, (short)0, c1, false, false);
    c2 = __builtin_amdgcn_wmma_f32_16x16x32_f16(false, af, false, b2, (short)0, c2, false, false);
    c3 = __builtin_amdgcn_wmma_f32_16x16x32_f16(false, af, false, b3, (short)0, c3, false, false);
  }

  float dsum = (ds0 + ds1) + (ds2 + ds3);

  // Each row's denominator is split across lane L and lane L+16 (disjoint K
  // offsets covering all 32 s per chunk) -> merge, broadcast via LDS.
  dsum += __shfl_xor(dsum, 16);
  if (lane < 16) denom_lds[lane] = dsum;
  __syncthreads();

  float invd[8];
#pragma unroll
  for (int i = 0; i < 8; ++i) invd[i] = 1.0f / denom_lds[i + hi * 8];

  // C/D layout: VGPR i -> row i (lanes 0-15) or row i+8 (lanes 16-31), N=lane%16
  float* outp = out + ((size_t)bc * T_DIM) * K_DIM;
#pragma unroll
  for (int i = 0; i < 8; ++i) {
    int row = tbase + i + hi * 8;
    float* o = outp + (size_t)row * K_DIM + lrow;
    o[0]  = c0[i] * invd[i];
    o[16] = c1[i] * invd[i];
    o[32] = c2[i] * invd[i];
    o[48] = c3[i] * invd[i];
  }
}

// ---------------------------------------------------------------------------
extern "C" void kernel_launch(void* const* d_in, const int* in_sizes, int n_in,
                              void* d_out, int out_size, void* d_ws, size_t ws_size,
                              hipStream_t stream) {
  const float* x  = (const float*)d_in[0];
  const float* Wq = (const float*)d_in[1];
  const float* bq = (const float*)d_in[2];
  const float* Wk = (const float*)d_in[3];
  const float* bk = (const float*)d_in[4];
  const float* Wv = (const float*)d_in[5];
  const float* bv = (const float*)d_in[6];
  float* out = (float*)d_out;

  // Workspace layout:
  //   [0, 256K)       a_arr  = scale*qs, 65536 f32
  //   [256K, 512K)    ks_arr = ks,       65536 f32
  //   [512K, 1M)      misc (weight sums, bias sums, ks max/min)
  //   [1M, 9M)        Vt: f16 V transposed [bc][d][t], 32*64*2048 halves
  char* ws = (char*)d_ws;
  float*    a_arr  = (float*)ws;
  float*    ks_arr = (float*)(ws + 262144);
  float*    misc   = (float*)(ws + 524288);
  _Float16* Vt     = (_Float16*)(ws + 1048576);

  prep_weights<<<1, 64, 0, stream>>>(Wq, bq, Wk, bk, misc);
  proj_kernel<<<1024, 64, 0, stream>>>(x, Wv, bv, misc, a_arr, ks_arr, Vt);
  minmax_kernel<<<32, 256, 0, stream>>>(ks_arr, misc);
  attn_kernel<<<dim3(128, 32), 32, 0, stream>>>(a_arr, ks_arr, misc, Vt, out);
}
